// CascadeRefineHead_27324581937579
// MI455X (gfx1250) — compile-verified, matrix operand order checked by reading, not run on hardware
//
#include <hip/hip_runtime.h>
#include <hip/hip_bf16.h>

typedef __attribute__((ext_vector_type(16))) _Float16 v16h;
typedef __attribute__((ext_vector_type(8)))  _Float16 v8h;
typedef __attribute__((ext_vector_type(8)))  float    v8f;

#define NUM_POINTS    72
#define SAMPLE_POINTS 36
#define C_FEAT        64
#define HIDDEN        192
#define GROUPS        6
#define HEAD_DIM      32
#define N_PRIORS      192
#define BATCH         64
#define ROWS          (BATCH * N_PRIORS)          /* 12288 */
#define FEATK         (SAMPLE_POINTS * C_FEAT)    /* 2304  */
#define PRIOR_W       (4 + NUM_POINTS)            /* 76    */
#define OUT_W         (NUM_POINTS + 4)            /* 76    */
#define REG_W         (NUM_POINTS + 2)            /* 74    */

// ---------------------------------------------------------------------------
// channels-last copy: src (B,64,HW) -> dst (B,HW,64) for coalesced sampling
// ---------------------------------------------------------------------------
__global__ void to_cl_kernel(const float* __restrict__ src, float* __restrict__ dst, int HW) {
    int idx = blockIdx.x * blockDim.x + threadIdx.x;
    int total = BATCH * HW * C_FEAT;
    if (idx >= total) return;
    int c = idx & 63;
    int p = (idx >> 6) % HW;
    int b = idx / (HW * C_FEAT);
    dst[idx] = src[((size_t)(b * C_FEAT + c)) * HW + p];
}

// ---------------------------------------------------------------------------
// weight prep: f32 [K][O] -> f16 [Opad][K] (transposed, zero-padded rows)
// ---------------------------------------------------------------------------
__global__ void transpose_pad_f16(const float* __restrict__ src, _Float16* __restrict__ dst,
                                  int K, int O, int Opad) {
    int idx = blockIdx.x * blockDim.x + threadIdx.x;
    if (idx >= K * Opad) return;
    int o = idx / K, k = idx - o * K;
    dst[idx] = (o < O) ? (_Float16)src[(size_t)k * O + o] : (_Float16)0.f;
}

__global__ void convert_f16(const float* __restrict__ src, _Float16* __restrict__ dst, int n) {
    int idx = blockIdx.x * blockDim.x + threadIdx.x;
    if (idx < n) dst[idx] = (_Float16)src[idx];
}

// ---------------------------------------------------------------------------
// multi-level bilinear sampling with z-softmax weighting
// grid = (S, N_PRIORS, BATCH), block = 64 (one thread per channel)
// ---------------------------------------------------------------------------
__device__ __forceinline__ float bilin_cl(const float* __restrict__ f, int b, int c,
                                          int H, int W, float px, float py) {
    float fx = floorf(px), fy = floorf(py);
    int x0 = (int)fx, y0 = (int)fy;
    float wx = px - fx, wy = py - fy;
    float v = 0.f;
#pragma unroll
    for (int dy = 0; dy < 2; ++dy) {
#pragma unroll
        for (int dx = 0; dx < 2; ++dx) {
            int xx = x0 + dx, yy = y0 + dy;
            if (xx >= 0 && xx < W && yy >= 0 && yy < H) {
                float w = (dx ? wx : 1.f - wx) * (dy ? wy : 1.f - wy);
                v += w * f[(((size_t)b * H + yy) * W + xx) * C_FEAT + c];
            }
        }
    }
    return v;
}

__global__ void __launch_bounds__(64)
sample_kernel(const float* __restrict__ f0, const float* __restrict__ f1,
              const float* __restrict__ f2, const float* __restrict__ priors,
              const float* __restrict__ z_emb, _Float16* __restrict__ feature) {
    int s = blockIdx.x, n = blockIdx.y, b = blockIdx.z, c = threadIdx.x;
    // flipped sample index: SAMPLE_X_IDX[35-s] = floor(71*(35-s)/35)
    int xidx = (71 * (35 - s)) / 35;
    float xs = priors[((size_t)(b * N_PRIORS + n)) * PRIOR_W + 4 + xidx];
    float ys = 1.0f - (float)xidx * (1.0f / 71.0f);
    // z softmax over 3 FPN levels (TAU = 2)
    float z  = z_emb[s];
    float e0 = -(z - 0.f) * (z - 0.f) * 0.5f;
    float e1 = -(z - 1.f) * (z - 1.f) * 0.5f;
    float e2 = -(z - 2.f) * (z - 2.f) * 0.5f;
    float m  = fmaxf(e0, fmaxf(e1, e2));
    float w0 = __expf(e0 - m), w1 = __expf(e1 - m), w2 = __expf(e2 - m);
    float inv = 1.0f / (w0 + w1 + w2);
    w0 *= inv; w1 *= inv; w2 *= inv;
    // align_corners=True: x = xs*(W-1), y = ys*(H-1)
    float acc = 0.f;
    acc += w0 * bilin_cl(f0, b, c, 10,  25, xs * 24.f, ys * 9.f);
    acc += w1 * bilin_cl(f1, b, c, 20,  50, xs * 49.f, ys * 19.f);
    acc += w2 * bilin_cl(f2, b, c, 40, 100, xs * 99.f, ys * 39.f);
    feature[((size_t)(b * N_PRIORS + n)) * FEATK + s * C_FEAT + c] = (_Float16)acc;
}

// ---------------------------------------------------------------------------
// Register-blocked batched WMMA GEMM:
//   C[m][n] (+= R) = sum_k A[m][k] * Bt[n][k] (+bias), A/Bt f16.
// Each wave computes an MT x NT grid of 16x16 tiles with
// v_wmma_f32_16x16x32_f16, reusing A fragments across NT tiles and B
// fragments across MT tiles (MT*NT WMMAs per MT+NT fragment loads).
// Two-level batch (b1 outer, b2 = z % nb2) with element strides.
// K % 32 == 0, M % 16 == 0; ragged N handled via zero-padded Bt rows +
// store guard. blockDim = (32, 4): threadIdx.y selects the wave.
// ---------------------------------------------------------------------------
template <int MT, int NT>
__global__ void __launch_bounds__(128)
wmma_gemm(const _Float16* __restrict__ A, int lda, int sA1, int sA2,
          const _Float16* __restrict__ Bt, int ldb, int sB1, int sB2,
          const float* __restrict__ bias, int sBias2,
          const float* __restrict__ R, int sR1, int sR2,
          float* __restrict__ Cf, _Float16* __restrict__ Ch,
          int ldc, int sC1, int sC2,
          int M, int N, int K, int nb2, int relu) {
    int tmBase = (blockIdx.x * 4 + threadIdx.y) * MT;
    if (tmBase * 16 >= M) return;
    int tnBase = blockIdx.y * NT;
    int z    = blockIdx.z;
    int b1   = z / nb2, b2 = z - b1 * nb2;
    int lane = threadIdx.x;
    int lrow = lane & 15;
    int koff = (lane >> 4) * 8;

    const _Float16* abase = A  + (size_t)b1 * sA1 + (size_t)b2 * sA2 + koff;
    const _Float16* bbase = Bt + (size_t)b1 * sB1 + (size_t)b2 * sB2 + koff;
    const _Float16* ap[MT];
    const _Float16* bp[NT];
#pragma unroll
    for (int i = 0; i < MT; ++i)
        ap[i] = abase + (size_t)((tmBase + i) * 16 + lrow) * lda;
#pragma unroll
    for (int j = 0; j < NT; ++j)
        bp[j] = bbase + (size_t)((tnBase + j) * 16 + lrow) * ldb;

    v8f acc[MT][NT] = {};
    for (int k0 = 0; k0 < K; k0 += 32) {
        v16h a[MT], b[NT];
#pragma unroll
        for (int i = 0; i < MT; ++i) {
            __builtin_prefetch(ap[i] + k0 + 128, 0, 1);   // global_prefetch_b8
            v8h lo = *(const v8h*)(ap[i] + k0);
            v8h hi = *(const v8h*)(ap[i] + k0 + 16);
            a[i] = __builtin_shufflevector(lo, hi, 0,1,2,3,4,5,6,7,8,9,10,11,12,13,14,15);
        }
#pragma unroll
        for (int j = 0; j < NT; ++j) {
            __builtin_prefetch(bp[j] + k0 + 128, 0, 1);
            v8h lo = *(const v8h*)(bp[j] + k0);
            v8h hi = *(const v8h*)(bp[j] + k0 + 16);
            b[j] = __builtin_shufflevector(lo, hi, 0,1,2,3,4,5,6,7,8,9,10,11,12,13,14,15);
        }
#pragma unroll
        for (int i = 0; i < MT; ++i)
#pragma unroll
            for (int j = 0; j < NT; ++j)
                acc[i][j] = __builtin_amdgcn_wmma_f32_16x16x32_f16(
                    false, a[i], false, b[j], (short)0, acc[i][j], false, false);
    }
    // C/D layout: VGPR e -> row tile*16 + (lane>>4)*8 + e, col = tile*16 + (lane&15)
    size_t cbase = (size_t)b1 * sC1 + (size_t)b2 * sC2;
    size_t rbase = (size_t)b1 * sR1 + (size_t)b2 * sR2;
#pragma unroll
    for (int j = 0; j < NT; ++j) {
        int col = (tnBase + j) * 16 + lrow;
        if (col >= N) continue;
        float bv = bias ? bias[(size_t)b2 * sBias2 + col] : 0.f;
#pragma unroll
        for (int i = 0; i < MT; ++i) {
            int m0 = (tmBase + i) * 16 + (lane >> 4) * 8;
#pragma unroll
            for (int e = 0; e < 8; ++e) {
                float v = acc[i][j][e] + bv;
                size_t mi = (size_t)(m0 + e);
                if (R) v += R[rbase + mi * ldc + col];
                if (relu) v = fmaxf(v, 0.f);
                if (Cf) Cf[cbase + mi * ldc + col] = v;
                if (Ch) Ch[cbase + mi * ldc + col] = (_Float16)v;
            }
        }
    }
}

// ---------------------------------------------------------------------------
// kxT[b][g][d][m] = feat16[b*192+m][g*32+d]  (B for ctx GEMM)
// ---------------------------------------------------------------------------
__global__ void kx_transpose(const _Float16* __restrict__ feat16, _Float16* __restrict__ kxT) {
    int idx = blockIdx.x * blockDim.x + threadIdx.x;
    if (idx >= BATCH * GROUPS * HEAD_DIM * N_PRIORS) return;
    int m = idx % N_PRIORS;
    int d = (idx / N_PRIORS) % HEAD_DIM;
    int g = (idx / (N_PRIORS * HEAD_DIM)) % GROUPS;
    int b = idx / (N_PRIORS * HEAD_DIM * GROUPS);
    kxT[idx] = feat16[((size_t)(b * N_PRIORS + m)) * HIDDEN + g * HEAD_DIM + d];
}

// ---------------------------------------------------------------------------
// row softmax over 192 cols with scale, f32 -> f16. One wave32 per row.
// blockDim = 256 (8 waves); rows must be a multiple of 8 per grid config.
// ---------------------------------------------------------------------------
__global__ void __launch_bounds__(256)
softmax_kernel(const float* __restrict__ S, _Float16* __restrict__ attn, float scale) {
    int row  = blockIdx.x * 8 + (threadIdx.x >> 5);
    int lane = threadIdx.x & 31;
    const float* sp = S + (size_t)row * N_PRIORS;
    float v[6];
    float mx = -1e30f;
#pragma unroll
    for (int j = 0; j < 6; ++j) {
        v[j] = sp[lane + j * 32] * scale;
        mx = fmaxf(mx, v[j]);
    }
#pragma unroll
    for (int off = 16; off >= 1; off >>= 1)
        mx = fmaxf(mx, __shfl_xor(mx, off, 32));
    float sum = 0.f;
#pragma unroll
    for (int j = 0; j < 6; ++j) { v[j] = __expf(v[j] - mx); sum += v[j]; }
#pragma unroll
    for (int off = 16; off >= 1; off >>= 1)
        sum += __shfl_xor(sum, off, 32);
    float inv = 1.0f / sum;
    _Float16* op = attn + (size_t)row * N_PRIORS;
#pragma unroll
    for (int j = 0; j < 6; ++j) op[lane + j * 32] = (_Float16)(v[j] * inv);
}

// ---------------------------------------------------------------------------
// predictions[row][0:2] = cls ; [2:76] = priors[row][2:76] + reg[row][0:74]
// ---------------------------------------------------------------------------
__global__ void concat_kernel(const float* __restrict__ cls32, const float* __restrict__ reg32,
                              const float* __restrict__ priors, float* __restrict__ out) {
    int idx = blockIdx.x * blockDim.x + threadIdx.x;
    if (idx >= ROWS * OUT_W) return;
    int row = idx / OUT_W, j = idx - row * OUT_W;
    float v;
    if (j < 2) v = cls32[(size_t)row * 2 + j];
    else       v = priors[(size_t)row * PRIOR_W + j] + reg32[(size_t)row * REG_W + (j - 2)];
    out[idx] = v;
}

// ---------------------------------------------------------------------------
extern "C" void kernel_launch(void* const* d_in, const int* in_sizes, int n_in,
                              void* d_out, int out_size, void* d_ws, size_t ws_size,
                              hipStream_t stream) {
    const float* feat0    = (const float*)d_in[0];
    const float* feat1    = (const float*)d_in[1];
    const float* feat2    = (const float*)d_in[2];
    const float* priors   = (const float*)d_in[3];
    const float* z_emb    = (const float*)d_in[4];
    const float* gather_w = (const float*)d_in[5];
    const float* gather_b = (const float*)d_in[6];
    const float* q_w      = (const float*)d_in[7];
    const float* k_w      = (const float*)d_in[8];
    const float* ch_w1    = (const float*)d_in[9];
    const float* ch_b1    = (const float*)d_in[10];
    const float* ch_w2    = (const float*)d_in[11];
    const float* ch_b2    = (const float*)d_in[12];
    const float* reg_m_w  = (const float*)d_in[13];
    const float* reg_m_b  = (const float*)d_in[14];
    const float* cls_m_w  = (const float*)d_in[15];
    const float* cls_m_b  = (const float*)d_in[16];
    const float* reg_w    = (const float*)d_in[17];
    const float* reg_b    = (const float*)d_in[18];
    const float* cls_w    = (const float*)d_in[19];
    const float* cls_b    = (const float*)d_in[20];
    float* out = (float*)d_out;

    // ---- workspace carve-up (256B aligned) ----
    char*  ws  = (char*)d_ws;
    size_t off = 0;
    auto carve = [&](size_t bytes) -> char* {
        char* p = ws + off;
        off = (off + bytes + 255) & ~(size_t)255;
        return p;
    };
    float*    f0cl    = (float*)   carve((size_t)BATCH * 250  * C_FEAT * 4);
    float*    f1cl    = (float*)   carve((size_t)BATCH * 1000 * C_FEAT * 4);
    float*    f2cl    = (float*)   carve((size_t)BATCH * 4000 * C_FEAT * 4);
    _Float16* feature = (_Float16*)carve((size_t)ROWS * FEATK * 2);
    _Float16* Wg16    = (_Float16*)carve((size_t)HIDDEN * 384 * 2);
    _Float16* Wq_t    = (_Float16*)carve((size_t)HIDDEN * HIDDEN * 2);
    _Float16* Wk_t    = (_Float16*)carve((size_t)HEAD_DIM * HEAD_DIM * 2);
    _Float16* Wch1_t  = (_Float16*)carve((size_t)384 * HIDDEN * 2);
    _Float16* Wch2_t  = (_Float16*)carve((size_t)HIDDEN * 384 * 2);
    _Float16* Wclsm_t = (_Float16*)carve((size_t)HIDDEN * HIDDEN * 2);
    _Float16* Wregm_t = (_Float16*)carve((size_t)HIDDEN * HIDDEN * 2);
    _Float16* Wcls_t  = (_Float16*)carve((size_t)16 * HIDDEN * 2);
    _Float16* Wreg_t  = (_Float16*)carve((size_t)80 * HIDDEN * 2);
    float*    feat32  = (float*)   carve((size_t)ROWS * HIDDEN * 4);
    _Float16* feat16  = (_Float16*)carve((size_t)ROWS * HIDDEN * 2);
    _Float16* q16     = (_Float16*)carve((size_t)ROWS * HIDDEN * 2);
    _Float16* k16     = (_Float16*)carve((size_t)ROWS * HIDDEN * 2);   // (B,G,192,32)
    _Float16* kxT16   = (_Float16*)carve((size_t)ROWS * HIDDEN * 2);   // (B,G,32,192)
    float*    S32     = (float*)   carve((size_t)BATCH * GROUPS * N_PRIORS * N_PRIORS * 4);
    _Float16* attn16  = (_Float16*)carve((size_t)BATCH * GROUPS * N_PRIORS * N_PRIORS * 2);
    _Float16* ctx16   = (_Float16*)carve((size_t)ROWS * HIDDEN * 2);
    _Float16* h1_16   = (_Float16*)carve((size_t)ROWS * 384 * 2);
    _Float16* featR16 = (_Float16*)carve((size_t)ROWS * HIDDEN * 2);
    _Float16* clsf16  = (_Float16*)carve((size_t)ROWS * HIDDEN * 2);
    _Float16* regf16  = (_Float16*)carve((size_t)ROWS * HIDDEN * 2);
    float*    cls32   = (float*)   carve((size_t)ROWS * 2 * 4);
    float*    reg32   = (float*)   carve((size_t)ROWS * REG_W * 4);
    (void)ws_size; (void)in_sizes; (void)n_in; (void)out_size;

    // ---- 1. channels-last feature maps ----
    to_cl_kernel<<<(BATCH*250 *C_FEAT + 255)/256, 256, 0, stream>>>(feat0, f0cl, 250);
    to_cl_kernel<<<(BATCH*1000*C_FEAT + 255)/256, 256, 0, stream>>>(feat1, f1cl, 1000);
    to_cl_kernel<<<(BATCH*4000*C_FEAT + 255)/256, 256, 0, stream>>>(feat2, f2cl, 4000);

    // ---- 2. weight prep (f16, [out][in] layout, padded) ----
    convert_f16      <<<(HIDDEN*384 + 255)/256, 256, 0, stream>>>(gather_w, Wg16, HIDDEN*384);
    transpose_pad_f16<<<(HIDDEN*HIDDEN + 255)/256, 256, 0, stream>>>(q_w,     Wq_t,    HIDDEN, HIDDEN, HIDDEN);
    transpose_pad_f16<<<(HEAD_DIM*HEAD_DIM + 255)/256, 256, 0, stream>>>(k_w, Wk_t,    HEAD_DIM, HEAD_DIM, HEAD_DIM);
    transpose_pad_f16<<<(HIDDEN*384 + 255)/256, 256, 0, stream>>>(ch_w1,   Wch1_t,  HIDDEN, 384, 384);
    transpose_pad_f16<<<(384*HIDDEN + 255)/256, 256, 0, stream>>>(ch_w2,   Wch2_t,  384, HIDDEN, HIDDEN);
    transpose_pad_f16<<<(HIDDEN*HIDDEN + 255)/256, 256, 0, stream>>>(cls_m_w, Wclsm_t, HIDDEN, HIDDEN, HIDDEN);
    transpose_pad_f16<<<(HIDDEN*HIDDEN + 255)/256, 256, 0, stream>>>(reg_m_w, Wregm_t, HIDDEN, HIDDEN, HIDDEN);
    transpose_pad_f16<<<(HIDDEN*16 + 255)/256, 256, 0, stream>>>(cls_w,   Wcls_t,  HIDDEN, 2, 16);
    transpose_pad_f16<<<(HIDDEN*80 + 255)/256, 256, 0, stream>>>(reg_w,   Wreg_t,  HIDDEN, REG_W, 80);

    // ---- 3. z-weighted multi-level bilinear sampling -> feature (f16) ----
    sample_kernel<<<dim3(SAMPLE_POINTS, N_PRIORS, BATCH), 64, 0, stream>>>(
        f0cl, f1cl, f2cl, priors, z_emb, feature);

    dim3 blk(32, 4);
    // ---- 4. gather_fc (grouped): 6x [12288 x 32 x 384] -> feat (f32 + f16) ----
    wmma_gemm<2,2><<<dim3(96, 1, GROUPS), blk, 0, stream>>>(
        feature, FEATK, 0, 384,
        Wg16, 384, 0, 32*384,
        gather_b, 32,
        nullptr, 0, 0,
        feat32, feat16, HIDDEN, 0, 32,
        ROWS, 32, 384, GROUPS, 0);

    // ---- 5. q = feat @ q_w : [12288 x 192 x 192] (f16) ----
    wmma_gemm<2,4><<<dim3(96, 3, 1), blk, 0, stream>>>(
        feat16, HIDDEN, 0, 0,
        Wq_t, HIDDEN, 0, 0,
        nullptr, 0, nullptr, 0, 0,
        nullptr, q16, HIDDEN, 0, 0,
        ROWS, HIDDEN, HIDDEN, 1, 0);

    // ---- 6. k = kx @ k_w : per (b,g) [192 x 32 x 32] -> k16 (B,G,192,32) ----
    wmma_gemm<2,2><<<dim3(2, 1, BATCH*GROUPS), blk, 0, stream>>>(
        feat16, HIDDEN, N_PRIORS*HIDDEN, HEAD_DIM,
        Wk_t, HEAD_DIM, 0, 0,
        nullptr, 0, nullptr, 0, 0,
        nullptr, k16, HEAD_DIM, GROUPS*N_PRIORS*HEAD_DIM, N_PRIORS*HEAD_DIM,
        N_PRIORS, HEAD_DIM, HEAD_DIM, GROUPS, 0);

    // ---- 7. kxT (B,G,32,192) for ctx GEMM ----
    kx_transpose<<<(ROWS*HIDDEN + 255)/256, 256, 0, stream>>>(feat16, kxT16);

    // ---- 8. S = q . k^T : per (b,g) [192 x 192 x 32] (f32) ----
    wmma_gemm<2,4><<<dim3(2, 3, BATCH*GROUPS), blk, 0, stream>>>(
        q16, HIDDEN, N_PRIORS*HIDDEN, HEAD_DIM,
        k16, HEAD_DIM, GROUPS*N_PRIORS*HEAD_DIM, N_PRIORS*HEAD_DIM,
        nullptr, 0, nullptr, 0, 0,
        S32, nullptr, N_PRIORS, GROUPS*N_PRIORS*N_PRIORS, N_PRIORS*N_PRIORS,
        N_PRIORS, N_PRIORS, HEAD_DIM, GROUPS, 0);

    // ---- 9. softmax(S * hd^-0.5) -> attn (f16) ----
    softmax_kernel<<<(BATCH*GROUPS*N_PRIORS)/8, 256, 0, stream>>>(
        S32, attn16, 0.17677669529663687f /* 32^-0.5 */);

    // ---- 10. ctx = attn @ kx : per (b,g) [192 x 32 x 192] -> ctx16 (B,N,192) ----
    wmma_gemm<2,2><<<dim3(2, 1, BATCH*GROUPS), blk, 0, stream>>>(
        attn16, N_PRIORS, GROUPS*N_PRIORS*N_PRIORS, N_PRIORS*N_PRIORS,
        kxT16, N_PRIORS, GROUPS*HEAD_DIM*N_PRIORS, HEAD_DIM*N_PRIORS,
        nullptr, 0, nullptr, 0, 0,
        nullptr, ctx16, HIDDEN, N_PRIORS*HIDDEN, HEAD_DIM,
        N_PRIORS, HEAD_DIM, HIDDEN, GROUPS, 0);

    // ---- 11. h1 = relu(ctx @ ch_w1 + b1) : [12288 x 384 x 192] ----
    wmma_gemm<2,4><<<dim3(96, 6, 1), blk, 0, stream>>>(
        ctx16, HIDDEN, 0, 0,
        Wch1_t, HIDDEN, 0, 0,
        ch_b1, 0, nullptr, 0, 0,
        nullptr, h1_16, 384, 0, 0,
        ROWS, 384, HIDDEN, 1, 1);

    // ---- 12. featR = feat + h1 @ ch_w2 + b2 : [12288 x 192 x 384] ----
    wmma_gemm<2,4><<<dim3(96, 3, 1), blk, 0, stream>>>(
        h1_16, 384, 0, 0,
        Wch2_t, 384, 0, 0,
        ch_b2, 0, feat32, 0, 0,
        nullptr, featR16, HIDDEN, 0, 0,
        ROWS, HIDDEN, 384, 1, 0);

    // ---- 13. cls_f = relu(featR @ cls_m_w + b) ----
    wmma_gemm<2,4><<<dim3(96, 3, 1), blk, 0, stream>>>(
        featR16, HIDDEN, 0, 0,
        Wclsm_t, HIDDEN, 0, 0,
        cls_m_b, 0, nullptr, 0, 0,
        nullptr, clsf16, HIDDEN, 0, 0,
        ROWS, HIDDEN, HIDDEN, 1, 1);

    // ---- 14. reg_f = relu(featR @ reg_m_w + b) ----
    wmma_gemm<2,4><<<dim3(96, 3, 1), blk, 0, stream>>>(
        featR16, HIDDEN, 0, 0,
        Wregm_t, HIDDEN, 0, 0,
        reg_m_b, 0, nullptr, 0, 0,
        nullptr, regf16, HIDDEN, 0, 0,
        ROWS, HIDDEN, HIDDEN, 1, 1);

    // ---- 15. cls_logits = cls_f @ cls_w + b : N=2 (padded tile) ----
    wmma_gemm<2,1><<<dim3(96, 1, 1), blk, 0, stream>>>(
        clsf16, HIDDEN, 0, 0,
        Wcls_t, HIDDEN, 0, 0,
        cls_b, 0, nullptr, 0, 0,
        cls32, nullptr, 2, 0, 0,
        ROWS, 2, HIDDEN, 1, 0);

    // ---- 16. reg = reg_f @ reg_w + b : N=74 (padded to 80) ----
    wmma_gemm<2,1><<<dim3(96, 5, 1), blk, 0, stream>>>(
        regf16, HIDDEN, 0, 0,
        Wreg_t, HIDDEN, 0, 0,
        reg_b, 0, nullptr, 0, 0,
        reg32, nullptr, REG_W, 0, 0,
        ROWS, REG_W, HIDDEN, 1, 0);

    // ---- 17. predictions = concat([cls, priors[...,2:] + reg]) ----
    concat_kernel<<<(ROWS*OUT_W + 255)/256, 256, 0, stream>>>(cls32, reg32, priors, out);
}
